// GPTOSSAttentionBlock_27547920237063
// MI455X (gfx1250) — matrix operand order repaired and checked by reading, not compiled, not run on hardware
//
#include <hip/hip_runtime.h>
#include <hip/hip_bf16.h>
#include <math.h>

#define N_TOKENS 2048
#define HIDDEN   2880
#define HEAD_DIM 64
#define N_HEADS  64
#define N_KV     8
#define Q_MULT   8
#define WINDOW   128
#define QKV_DIM  5120          // 64*(64+16)
#define Q_DIM    4096          // N_HEADS*HEAD_DIM
#define K_OFF    4096
#define V_OFF    4608
#define SM_SCALE 0.125f        // 1/sqrt(64)
#define RMS_EPS  1e-5f

typedef __attribute__((ext_vector_type(16))) __bf16 v16bf;
typedef __attribute__((ext_vector_type(8)))  __bf16 v8bf;
typedef __attribute__((ext_vector_type(8)))  float  v8f;

union Frag { v16bf v; v8bf h[2]; };

__device__ __forceinline__ unsigned short f2bf(float f) {
  union { float f; unsigned u; } v; v.f = f;
  unsigned r = v.u + 0x7FFFu + ((v.u >> 16) & 1u);   // round-to-nearest-even
  return (unsigned short)(r >> 16);
}
__device__ __forceinline__ float bf2f(unsigned short u) {
  union { unsigned u; float f; } v; v.u = ((unsigned)u) << 16; return v.f;
}
__device__ __forceinline__ v8bf ld8(const unsigned short* p) {
  return *reinterpret_cast<const v8bf*>(p);
}
__device__ __forceinline__ v8f wmma_bf16(v16bf a, v16bf b, v8f c) {
  // D(f32) = A(16x32 bf16) * B(32x16 bf16) + C
  return __builtin_amdgcn_wmma_f32_16x16x32_bf16(false, a, false, b, (short)0, c,
                                                 false, false);
}

// ---------------- f32 -> bf16 convert ----------------
__global__ void k_f32_to_bf16(const float* __restrict__ src,
                              unsigned short* __restrict__ dst, int n) {
  int i = blockIdx.x * blockDim.x + threadIdx.x;
  int stride = gridDim.x * blockDim.x;
  for (; i < n; i += stride) dst[i] = f2bf(src[i]);
}

// ---------------- RMSNorm -> bf16 ----------------
__global__ void __launch_bounds__(256) k_rmsnorm(const float* __restrict__ x,
                                                 const float* __restrict__ scale,
                                                 unsigned short* __restrict__ t) {
  __shared__ float red[256];
  int row = blockIdx.x;
  const float* xr = x + (size_t)row * HIDDEN;
  float s = 0.f;
  for (int c = threadIdx.x; c < HIDDEN; c += 256) { float v = xr[c]; s += v * v; }
  red[threadIdx.x] = s;
  __syncthreads();
  for (int off = 128; off > 0; off >>= 1) {
    if ((int)threadIdx.x < off) red[threadIdx.x] += red[threadIdx.x + off];
    __syncthreads();
  }
  float r = rsqrtf(red[0] / (float)HIDDEN + RMS_EPS);
  for (int c = threadIdx.x; c < HIDDEN; c += 256)
    t[(size_t)row * HIDDEN + c] = f2bf(xr[c] * r * scale[c]);
}

// ---------------- WMMA GEMM: C[M][N] = A[M][K] * B[N][K]^T + bias (+residual) ----
// block = 256 threads = 8 waves (4x2); block tile 128(M) x 64(N);
// wave tile 32x32 -> 2x2 register blocking: 8 b128 loads per 4 WMMAs, each
// fragment reused twice.
__global__ void __launch_bounds__(256) k_gemm_bf16(
    const unsigned short* __restrict__ A, const unsigned short* __restrict__ B,
    const float* __restrict__ bias, const float* __restrict__ residual,
    float* __restrict__ C, int N, int K) {
  int lane = threadIdx.x & 31;
  int wave = threadIdx.x >> 5;
  int lr = lane & 15;
  int g  = lane >> 4;
  int m0 = blockIdx.x * 128 + (wave >> 1) * 32;
  int n0 = blockIdx.y * 64  + (wave & 1) * 32;

  const unsigned short* a0 = A + (size_t)(m0 + lr) * K;
  const unsigned short* a1 = A + (size_t)(m0 + 16 + lr) * K;
  const unsigned short* b0 = B + (size_t)(n0 + lr) * K;
  const unsigned short* b1 = B + (size_t)(n0 + 16 + lr) * K;

  v8f acc00 = {}, acc01 = {}, acc10 = {}, acc11 = {};
  for (int k0 = 0; k0 < K; k0 += 32) {
    Frag fa0, fa1, fb0, fb1;
    fa0.h[0] = ld8(a0 + k0 + g * 8);
    fa0.h[1] = ld8(a0 + k0 + 16 + g * 8);
    fa1.h[0] = ld8(a1 + k0 + g * 8);
    fa1.h[1] = ld8(a1 + k0 + 16 + g * 8);
    fb0.h[0] = ld8(b0 + k0 + g * 8);
    fb0.h[1] = ld8(b0 + k0 + 16 + g * 8);
    fb1.h[0] = ld8(b1 + k0 + g * 8);
    fb1.h[1] = ld8(b1 + k0 + 16 + g * 8);
    acc00 = wmma_bf16(fa0.v, fb0.v, acc00);
    acc01 = wmma_bf16(fa0.v, fb1.v, acc01);
    acc10 = wmma_bf16(fa1.v, fb0.v, acc10);
    acc11 = wmma_bf16(fa1.v, fb1.v, acc11);
  }
  int col0 = n0 + lr;
  int col1 = n0 + 16 + lr;
  float bc0 = bias ? bias[col0] : 0.0f;
  float bc1 = bias ? bias[col1] : 0.0f;
#pragma unroll
  for (int r = 0; r < 8; ++r) {
    int row0 = m0 + r + g * 8;
    int row1 = row0 + 16;
    float v00 = acc00[r] + bc0;
    float v01 = acc01[r] + bc1;
    float v10 = acc10[r] + bc0;
    float v11 = acc11[r] + bc1;
    if (residual) {
      v00 += residual[(size_t)row0 * N + col0];
      v01 += residual[(size_t)row0 * N + col1];
      v10 += residual[(size_t)row1 * N + col0];
      v11 += residual[(size_t)row1 * N + col1];
    }
    C[(size_t)row0 * N + col0] = v00;
    C[(size_t)row0 * N + col1] = v01;
    C[(size_t)row1 * N + col0] = v10;
    C[(size_t)row1 * N + col1] = v11;
  }
}

// ---------------- RoPE (YaRN) split: qkv f32 -> Qbf/Kbf/Vt(bf16, transposed) ----
__device__ __forceinline__ void rope_cs(int f, float pos, float& c, float& s) {
  const float conc  = 0.1f * __logf(32.0f) + 1.0f;
  const float lg    = __logf(150000.0f);
  const float low   = 32.0f * __logf(4096.0f / (32.0f * 2.0f * (float)M_PI)) / lg;
  const float high  = 32.0f * __logf(4096.0f / (2.0f * (float)M_PI)) / lg;
  float freq   = __powf(150000.0f, (2.0f * (float)f) / 64.0f);
  float interp = 1.0f / (32.0f * freq);
  float extrap = 1.0f / freq;
  float ramp   = ((float)f - low) / (high - low);
  float mask   = 1.0f - fminf(fmaxf(ramp, 0.0f), 1.0f);
  float invf   = interp * (1.0f - mask) + extrap * mask;
  float ang    = pos * invf;
  c = cosf(ang) * conc;
  s = sinf(ang) * conc;
}

__global__ void __launch_bounds__(256) k_rope_split(
    const float* __restrict__ qkv, unsigned short* __restrict__ Qb,
    unsigned short* __restrict__ Kb, unsigned short* __restrict__ Vt) {
  int tok = blockIdx.x;
  const float* src = qkv + (size_t)tok * QKV_DIM;
  float pos = (float)tok;
  // Q: 64 heads x 32 freq pairs
  for (int p = threadIdx.x; p < N_HEADS * 32; p += 256) {
    int hgl = p >> 5, f = p & 31;
    float c, s; rope_cs(f, pos, c, s);
    float x1 = src[hgl * 64 + f];
    float x2 = src[hgl * 64 + f + 32];
    size_t base = ((size_t)tok * N_HEADS + hgl) * HEAD_DIM;
    Qb[base + f]      = f2bf(x1 * c - x2 * s);
    Qb[base + f + 32] = f2bf(x2 * c + x1 * s);
  }
  // K: 8 heads x 32 freq pairs
  for (int p = threadIdx.x; p < N_KV * 32; p += 256) {
    int h = p >> 5, f = p & 31;
    float c, s; rope_cs(f, pos, c, s);
    float x1 = src[K_OFF + h * 64 + f];
    float x2 = src[K_OFF + h * 64 + f + 32];
    size_t base = ((size_t)tok * N_KV + h) * HEAD_DIM;
    Kb[base + f]      = f2bf(x1 * c - x2 * s);
    Kb[base + f + 32] = f2bf(x2 * c + x1 * s);
  }
  // V: transpose to [h][d][token] so PV B-fragments load contiguously
  for (int e = threadIdx.x; e < N_KV * HEAD_DIM; e += 256) {
    int h = e >> 6, d = e & 63;
    Vt[((size_t)(h * 64 + d)) * N_TOKENS + tok] = f2bf(src[V_OFF + h * 64 + d]);
  }
}

// ---------------- Sliding-window attention with sinks ----------------
// grid = (128 query tiles, 8 kv heads); wave m handles q-head m.
__global__ void __launch_bounds__(256) k_attn(
    const unsigned short* __restrict__ Qb, const unsigned short* __restrict__ Kb,
    const unsigned short* __restrict__ Vt, const float* __restrict__ sinks,
    unsigned short* __restrict__ attn) {
  __shared__ alignas(16) float Sbuf[8][16][160];  // 80 KB; P(bf16) aliases rows in-place
  int qt   = blockIdx.x;
  int h    = blockIdx.y;
  int wave = threadIdx.x >> 5;
  int lane = threadIdx.x & 31;
  int lr   = lane & 15;
  int g    = lane >> 4;
  int hg   = h * Q_MULT + wave;
  int i0   = qt * 16;
  int jt_lo = (qt >= 8) ? (qt - 8) : 0;
  int numjt = qt - jt_lo + 1;            // <= 9 key tiles
  int ncols = numjt * 16;

  // Q A-fragments (two 32-wide d chunks)
  const unsigned short* qrow = Qb + (((size_t)(i0 + lr)) * N_HEADS + hg) * HEAD_DIM;
  Frag q0, q1;
  q0.h[0] = ld8(qrow + g * 8);       q0.h[1] = ld8(qrow + 16 + g * 8);
  q1.h[0] = ld8(qrow + 32 + g * 8);  q1.h[1] = ld8(qrow + 48 + g * 8);

  // S = Q K^T, masked + scaled, into LDS
  for (int t = 0; t < numjt; ++t) {
    int j0 = (jt_lo + t) * 16;
    const unsigned short* krow = Kb + (((size_t)(j0 + lr)) * N_KV + h) * HEAD_DIM;
    Frag b0, b1;
    b0.h[0] = ld8(krow + g * 8);       b0.h[1] = ld8(krow + 16 + g * 8);
    b1.h[0] = ld8(krow + 32 + g * 8);  b1.h[1] = ld8(krow + 48 + g * 8);
    v8f acc = {};
    acc = wmma_bf16(q0.v, b0.v, acc);
    acc = wmma_bf16(q1.v, b1.v, acc);
#pragma unroll
    for (int r = 0; r < 8; ++r) {
      int qg = i0 + r + g * 8;
      int jg = j0 + lr;
      bool ok = (jg <= qg) && (jg + WINDOW > qg);
      Sbuf[wave][r + g * 8][t * 16 + lr] = ok ? acc[r] * SM_SCALE : -INFINITY;
    }
  }
  __syncthreads();

  // Per-row softmax with sink logit; pack P(bf16) in-place at row start.
  if (lane < 16) {
    float* Srow = &Sbuf[wave][lr][0];
    unsigned short* Prow = reinterpret_cast<unsigned short*>(Srow);
    float snk = sinks[hg];
    float m = snk;
    for (int c = 0; c < ncols; ++c) m = fmaxf(m, Srow[c]);
    float sum = __expf(snk - m);
    for (int c = 0; c < ncols; ++c) {
      float e = __expf(Srow[c] - m);
      sum += e;
      Prow[c] = f2bf(e);               // overwrites Srow[c/2] low half (already consumed)
    }
    for (int c = ncols; c < 160; ++c) Prow[c] = 0;
    float inv = 1.0f / sum;
    for (int c = 0; c < ncols; ++c) Prow[c] = f2bf(bf2f(Prow[c]) * inv);
  }
  __syncthreads();

  // O = P V  (A = P from LDS, B = V from transposed global layout)
  int nsteps = (ncols + 31) >> 5;
  v8f o[4] = {};
  const unsigned short* Pbase = reinterpret_cast<const unsigned short*>(&Sbuf[wave][lr][0]);
  for (int ks = 0; ks < nsteps; ++ks) {
    int kk0 = ks * 32;
    Frag pa;
    pa.h[0] = *reinterpret_cast<const v8bf*>(Pbase + kk0 + g * 8);
    pa.h[1] = *reinterpret_cast<const v8bf*>(Pbase + kk0 + 16 + g * 8);
#pragma unroll
    for (int dt = 0; dt < 4; ++dt) {
      const unsigned short* vrow =
          Vt + ((size_t)(h * 64 + dt * 16 + lr)) * N_TOKENS + jt_lo * 16 + kk0 + g * 8;
      Frag vb;
      vb.h[0] = ld8(vrow);
      vb.h[1] = ld8(vrow + 16);
      o[dt] = wmma_bf16(pa.v, vb.v, o[dt]);
    }
  }
#pragma unroll
  for (int dt = 0; dt < 4; ++dt)
#pragma unroll
    for (int r = 0; r < 8; ++r) {
      int row = i0 + r + g * 8;
      int col = hg * 64 + dt * 16 + lr;
      attn[(size_t)row * Q_DIM + col] = f2bf(o[dt][r]);
    }
}

// ---------------- launch ----------------
extern "C" void kernel_launch(void* const* d_in, const int* in_sizes, int n_in,
                              void* d_out, int out_size, void* d_ws, size_t ws_size,
                              hipStream_t stream) {
  (void)in_sizes; (void)n_in; (void)out_size; (void)ws_size;
  const float* x     = (const float*)d_in[0];
  const float* nsc   = (const float*)d_in[1];
  const float* sinks = (const float*)d_in[2];
  const float* w_qkv = (const float*)d_in[3];
  const float* b_qkv = (const float*)d_in[4];
  const float* w_out = (const float*)d_in[5];
  const float* b_out = (const float*)d_in[6];
  float* out = (float*)d_out;

  char* ws = (char*)d_ws;
  size_t off = 0;
  auto alloc = [&](size_t bytes) -> void* {
    void* p = ws + off;
    off = (off + bytes + 255) & ~(size_t)255;
    return p;
  };
  unsigned short* t_bf    = (unsigned short*)alloc((size_t)N_TOKENS * HIDDEN * 2);
  unsigned short* wqkv_bf = (unsigned short*)alloc((size_t)QKV_DIM * HIDDEN * 2);
  unsigned short* wout_bf = (unsigned short*)alloc((size_t)HIDDEN * Q_DIM * 2);
  float*          qkv     = (float*)alloc((size_t)N_TOKENS * QKV_DIM * 4);
  unsigned short* Qb      = (unsigned short*)alloc((size_t)N_TOKENS * Q_DIM * 2);
  unsigned short* Kb      = (unsigned short*)alloc((size_t)N_TOKENS * N_KV * HEAD_DIM * 2);
  unsigned short* Vt      = (unsigned short*)alloc((size_t)N_KV * HEAD_DIM * N_TOKENS * 2 + 4096);
  unsigned short* attn    = (unsigned short*)alloc((size_t)N_TOKENS * Q_DIM * 2);

  k_f32_to_bf16<<<1024, 256, 0, stream>>>(w_qkv, wqkv_bf, QKV_DIM * HIDDEN);
  k_f32_to_bf16<<<1024, 256, 0, stream>>>(w_out, wout_bf, HIDDEN * Q_DIM);
  k_rmsnorm<<<N_TOKENS, 256, 0, stream>>>(x, nsc, t_bf);

  dim3 g1(N_TOKENS / 128, QKV_DIM / 64);
  k_gemm_bf16<<<g1, 256, 0, stream>>>(t_bf, wqkv_bf, b_qkv, nullptr, qkv, QKV_DIM, HIDDEN);

  k_rope_split<<<N_TOKENS, 256, 0, stream>>>(qkv, Qb, Kb, Vt);

  dim3 g2(N_TOKENS / 16, N_KV);
  k_attn<<<g2, 256, 0, stream>>>(Qb, Kb, Vt, sinks, attn);

  dim3 g3(N_TOKENS / 128, HIDDEN / 64);
  k_gemm_bf16<<<g3, 256, 0, stream>>>(attn, wout_bf, b_out, x, out, HIDDEN, Q_DIM);
}
